// BiGRU_43121471651933
// MI455X (gfx1250) — matrix operand order repaired and checked by reading, not compiled, not run on hardware
//
#include <hip/hip_runtime.h>
#include <hip/hip_bf16.h>

typedef __attribute__((ext_vector_type(16))) __bf16 v16bf;
typedef __attribute__((ext_vector_type(8)))  __bf16 v8bf;
typedef __attribute__((ext_vector_type(8)))  float  v8f;
typedef int v4i_ __attribute__((vector_size(16)));   // pointee type of async-LDS builtin

#define BATCH 32
#define SEQ   512
#define HID   512
#define G3H   1536   // 3*H

// ---------------------------------------------------------------------------
// CDNA5 async Global->LDS copy (ASYNCcnt path), with synchronous fallback.
// ---------------------------------------------------------------------------
#if __has_builtin(__builtin_amdgcn_global_load_async_to_lds_b128) && \
    __has_builtin(__builtin_amdgcn_s_wait_asynccnt)
#define HAS_ASYNC_LDS 1
#else
#define HAS_ASYNC_LDS 0
#endif

__device__ __forceinline__ void async_cp16(const __bf16* g, __bf16* l) {
#if HAS_ASYNC_LDS
  __builtin_amdgcn_global_load_async_to_lds_b128(
      (__attribute__((address_space(1))) v4i_*)g,
      (__attribute__((address_space(3))) v4i_*)l, 0, 0);
#else
  *(v8bf*)l = *(const v8bf*)g;
#endif
}

#if HAS_ASYNC_LDS
#define ASYNC_WAIT(n) __builtin_amdgcn_s_wait_asynccnt(n)
#else
#define ASYNC_WAIT(n) ((void)0)
#endif

// ---------------------------------------------------------------------------
// Fragment loaders (CDNA5 WMMA 16x16x32 bf16 layouts, ISA 7.12.2)
// ---------------------------------------------------------------------------

// A-matrix 16x32 bf16: lanes 0-15 rows M=0..15 (K=0..7 then 16..23),
// lanes 16-31 same rows (K=8..15 then 24..31).
__device__ __forceinline__ v16bf load_a_frag(const __bf16* base, int stride,
                                             int row0, int kk, int lane) {
  int r  = row0 + (lane & 15);
  int kb = kk + ((lane >> 4) ? 8 : 0);
  const __bf16* p = base + r * stride + kb;
  v8bf lo = *(const v8bf*)(p);
  v8bf hi = *(const v8bf*)(p + 16);
  v16bf a;
#pragma unroll
  for (int e = 0; e < 8; ++e) { a[e] = lo[e]; a[8 + e] = hi[e]; }
  return a;
}

// B-matrix 32x16 bf16: lane = column N, lanes 0-15 hold K=0..7 then 16..23,
// lanes 16-31 hold K=8..15 then 24..31.  Tile stored [N][K].
__device__ __forceinline__ v16bf load_b_frag(const __bf16* base, int stride,
                                             int col0, int kk, int lane) {
  int n  = col0 + (lane & 15);
  int kb = kk + ((lane >> 4) ? 8 : 0);
  const __bf16* p = base + n * stride + kb;
  v8bf lo = *(const v8bf*)(p);
  v8bf hi = *(const v8bf*)(p + 16);
  v16bf b;
#pragma unroll
  for (int e = 0; e < 8; ++e) { b[e] = lo[e]; b[8 + e] = hi[e]; }
  return b;
}

// Same layout, streamed straight from (L2-resident) bf16 weights in global.
__device__ __forceinline__ v16bf load_b_frag_g(const __bf16* __restrict__ w,
                                               int ldw, int col0, int kk, int lane) {
  int n  = col0 + (lane & 15);
  int kb = kk + ((lane >> 4) ? 8 : 0);
  const __bf16* p = w + (size_t)n * ldw + kb;
  v8bf lo = *(const v8bf*)(p);
  v8bf hi = *(const v8bf*)(p + 16);
  v16bf b;
#pragma unroll
  for (int e = 0; e < 8; ++e) { b[e] = lo[e]; b[8 + e] = hi[e]; }
  return b;
}

// ---------------------------------------------------------------------------
// fp32 -> bf16 conversion (weights / x), 4 elems per thread
// ---------------------------------------------------------------------------
__global__ __launch_bounds__(256) void cvt_bf16x4(const float* __restrict__ s,
                                                  __bf16* __restrict__ d, int n4) {
  int i = blockIdx.x * 256 + threadIdx.x;
  if (i < n4) {
    float4 v = *(const float4*)(s + (size_t)i * 4);
    __bf16* o = d + (size_t)i * 4;
    o[0] = (__bf16)v.x; o[1] = (__bf16)v.y; o[2] = (__bf16)v.z; o[3] = (__bf16)v.w;
  }
}

// ---------------------------------------------------------------------------
// Kernel 1: gx[s*B+b, :] = x[b,s,:] . wx^T + bx     (GX layout [S,B,3H])
// 128x64 tile / 256 threads, BK=32, double-buffered async Global->LDS staging.
// ---------------------------------------------------------------------------
#define KP 48                       // padded K stride (96B, b128-aligned)
__global__ __launch_bounds__(256) void gates_gemm(
    const __bf16* __restrict__ X, int K,
    const __bf16* __restrict__ W, const float* __restrict__ bias,
    float* __restrict__ GX) {
  __shared__ __bf16 As[2][128 * KP];
  __shared__ __bf16 Bs[2][64 * KP];
  const int tid  = threadIdx.x;
  const int lane = tid & 31;
  const int wave = tid >> 5;
  const int m0 = blockIdx.y * 128;       // over S*B
  const int n0 = blockIdx.x * 64;        // over 3H
  const int wc = wave & 3;               // N subtile
  const int wr = (wave >> 2) * 4;        // M subtile base (4 subtiles/wave)

  // staging assignments
  const int arow = tid >> 1;             // 0..127
  const int ac   = (tid & 1) * 16;       // bf16 col: 0 or 16 (two b128 each)
  const int brow = tid >> 2;             // 0..63
  const int bc   = (tid & 3) * 8;        // bf16 col: 0,8,16,24 (one b128)
  const int gm = m0 + arow;
  const int s_ = gm >> 5;                // / BATCH
  const int b_ = gm & 31;
  const __bf16* Ap = X + ((size_t)b_ * SEQ + s_) * K + ac;
  const __bf16* Wp = W + (size_t)(n0 + brow) * K + bc;
  __bf16* asl0 = &As[0][arow * KP + ac];
  __bf16* asl1 = &As[1][arow * KP + ac];
  __bf16* bsl0 = &Bs[0][brow * KP + bc];
  __bf16* bsl1 = &Bs[1][brow * KP + bc];

  v8f acc[4];
#pragma unroll
  for (int mt = 0; mt < 4; ++mt) acc[mt] = (v8f){};

  const int nb = K >> 5;  // K/32 stages
  // prologue: stage block 0 into buffer 0
  async_cp16(Ap, asl0);
  async_cp16(Ap + 8, asl0 + 8);
  async_cp16(Wp, bsl0);

  for (int i = 0; i < nb; ++i) {
    const int cur = i & 1;
    if (i + 1 < nb) {
      const int k0 = (i + 1) << 5;
      __bf16* al = (cur ? asl0 : asl1);
      __bf16* bl = (cur ? bsl0 : bsl1);
      async_cp16(Ap + k0, al);
      async_cp16(Ap + k0 + 8, al + 8);
      async_cp16(Wp + k0, bl);
      __builtin_prefetch(Ap + k0 + 32, 0, 0);   // global_prefetch_b8
      ASYNC_WAIT(3);                            // keep only just-issued outstanding
    } else {
      ASYNC_WAIT(0);
    }
    __syncthreads();                            // stage `i` visible to all waves

    const __bf16* at = &As[cur][0];
    const __bf16* bt = &Bs[cur][0];
    v16bf bf = load_b_frag(bt, KP, wc * 16, 0, lane);
#pragma unroll
    for (int mt = 0; mt < 4; ++mt) {
      v16bf af = load_a_frag(at, KP, (wr + mt) * 16, 0, lane);
      acc[mt] = __builtin_amdgcn_wmma_f32_16x16x32_bf16(false, af, false, bf,
                                                        (short)0, acc[mt], false, false);
    }
    __syncthreads();                            // done reading before overwrite
  }

  const int ncol = n0 + wc * 16 + (lane & 15);
  const float bv = bias[ncol];
#pragma unroll
  for (int mt = 0; mt < 4; ++mt) {
    const int mb = m0 + (wr + mt) * 16 + ((lane >> 4) * 8);
#pragma unroll
    for (int v = 0; v < 8; ++v)
      GX[(size_t)(mb + v) * G3H + ncol] = acc[mt][v] + bv;
  }
}

// ---------------------------------------------------------------------------
// Kernel 2: persistent GRU scan.  One 1024-thread block (32 waves) per
// direction; h (fp32 + bf16 mirror) and per-step gh live in LDS (~291KB of
// the 320KB WGP budget).  Per step: [32x1536] = h @ wh^T via bf16 WMMA
// (wh streamed bf16 from L2), then fused gates, barrier, next step.
// ---------------------------------------------------------------------------
#define HS32 520           // fp32 h row stride (pad)
#define HSBF 528           // bf16 h row stride (1056B = 66*16, b128-aligned)
#define GHS  1544          // gh fp32 row stride
#define OFF_HB  (BATCH * HS32 * 4)                 // 66560
#define OFF_GH  (OFF_HB + BATCH * HSBF * 2)        // 100352
#define SMEM_SCAN (OFF_GH + BATCH * GHS * 4)       // 297984 bytes

__global__ __launch_bounds__(1024) void gru_scan(
    const float* __restrict__ gxF, const float* __restrict__ gxB,
    const __bf16* __restrict__ whF, const float* __restrict__ bhF,
    const __bf16* __restrict__ whB, const float* __restrict__ bhB,
    float* __restrict__ Yf,         // fp32 [B,S,2H] (final layer) or null
    __bf16* __restrict__ Yb,        // bf16 [B,S,2H] (layer-0 feed) or null
    float* __restrict__ hNf, float* __restrict__ hNb) {
  extern __shared__ char smem[];
  float*  hS = (float*)smem;                  // [32][HS32] fp32 state
  __bf16* hB = (__bf16*)(smem + OFF_HB);      // [32][HSBF] bf16 mirror (WMMA A)
  float*  gh = (float*)(smem + OFF_GH);       // [32][GHS]  step gate scratch

  const int dir = blockIdx.x;
  const float*  gx = dir ? gxB : gxF;
  const __bf16* wh = dir ? whB : whF;
  const float*  bh = dir ? bhB : bhF;
  float* hN = dir ? hNb : hNf;

  const int tid  = threadIdx.x;
  const int lane = tid & 31;
  const int wave = tid >> 5;
  const int ncol0 = wave * 48;                // this wave's 48 gate columns

  for (int i = tid; i < BATCH * HS32; i += 1024) hS[i] = 0.f;
  for (int i = tid; i < BATCH * HSBF; i += 1024) hB[i] = (__bf16)0.f;
  __syncthreads();

  for (int t = 0; t < SEQ; ++t) {
    const int s = dir ? (SEQ - 1 - t) : t;

    // ---- gh = h @ wh^T  (wave owns rows 0..31 x cols [ncol0, ncol0+48)) ----
    v8f acc[2][3];
#pragma unroll
    for (int mt = 0; mt < 2; ++mt)
#pragma unroll
      for (int nt = 0; nt < 3; ++nt) acc[mt][nt] = (v8f){};

    for (int kk = 0; kk < HID; kk += 32) {
      v16bf a0 = load_a_frag(hB, HSBF, 0, kk, lane);
      v16bf a1 = load_a_frag(hB, HSBF, 16, kk, lane);
#pragma unroll
      for (int nt = 0; nt < 3; ++nt) {
        v16bf bf = load_b_frag_g(wh, HID, ncol0 + nt * 16, kk, lane);
        acc[0][nt] = __builtin_amdgcn_wmma_f32_16x16x32_bf16(false, a0, false, bf, (short)0, acc[0][nt], false, false);
        acc[1][nt] = __builtin_amdgcn_wmma_f32_16x16x32_bf16(false, a1, false, bf, (short)0, acc[1][nt], false, false);
      }
    }

    // spill gh tiles to LDS (C/D layout: M = v + 8*(lane>=16), N = lane&15)
#pragma unroll
    for (int mt = 0; mt < 2; ++mt)
#pragma unroll
      for (int nt = 0; nt < 3; ++nt) {
        int col   = ncol0 + nt * 16 + (lane & 15);
        int rbase = mt * 16 + ((lane >> 4) * 8);
#pragma unroll
        for (int v = 0; v < 8; ++v) gh[(rbase + v) * GHS + col] = acc[mt][nt][v];
      }
    __syncthreads();

    // ---- fused gates: r,z,n + state update (16 elements / thread) ----
    const float* gbase = gx + ((size_t)s * BATCH) * G3H;
    if (t + 1 < SEQ) {
      int snxt = dir ? (SEQ - 2 - t) : (t + 1);
      __builtin_prefetch(gx + ((size_t)snxt * BATCH) * G3H + tid * 48, 0, 0);
    }
#pragma unroll
    for (int e = 0; e < 16; ++e) {
      int ii = tid + e * 1024;
      int b = ii >> 9;
      int j = ii & 511;
      const float* g = gbase + (size_t)b * G3H;
      float ghr = gh[b * GHS + j]          + bh[j];
      float ghz = gh[b * GHS + HID + j]    + bh[HID + j];
      float ghn = gh[b * GHS + 2*HID + j]  + bh[2*HID + j];
      float r = 1.f / (1.f + __expf(-(g[j] + ghr)));
      float z = 1.f / (1.f + __expf(-(g[HID + j] + ghz)));
      float nx = g[2*HID + j] + r * ghn;
      float e2 = __expf(2.f * nx);
      float n = (e2 - 1.f) / (e2 + 1.f);          // tanh
      float hold = hS[b * HS32 + j];
      float hn = (1.f - z) * n + z * hold;
      hS[b * HS32 + j] = hn;
      hB[b * HSBF + j] = (__bf16)hn;
      size_t yi = ((size_t)b * SEQ + s) * (2 * HID) + dir * HID + j;
      if (Yf) Yf[yi] = hn;
      if (Yb) Yb[yi] = (__bf16)hn;
    }
    __syncthreads();
  }

  // final hidden state [B,H]
#pragma unroll
  for (int e = 0; e < 16; ++e) {
    int ii = tid + e * 1024;
    int b = ii >> 9;
    int j = ii & 511;
    hN[b * HID + j] = hS[b * HS32 + j];
  }
}

// ---------------------------------------------------------------------------
extern "C" void kernel_launch(void* const* d_in, const int* in_sizes, int n_in,
                              void* d_out, int out_size, void* d_ws, size_t ws_size,
                              hipStream_t stream) {
  const float* x     = (const float*)d_in[0];
  const float* wx_f0 = (const float*)d_in[1];
  const float* bx_f0 = (const float*)d_in[2];
  const float* wh_f0 = (const float*)d_in[3];
  const float* bh_f0 = (const float*)d_in[4];
  const float* wx_b0 = (const float*)d_in[5];
  const float* bx_b0 = (const float*)d_in[6];
  const float* wh_b0 = (const float*)d_in[7];
  const float* bh_b0 = (const float*)d_in[8];
  const float* wx_f1 = (const float*)d_in[9];
  const float* bx_f1 = (const float*)d_in[10];
  const float* wh_f1 = (const float*)d_in[11];
  const float* bh_f1 = (const float*)d_in[12];
  const float* wx_b1 = (const float*)d_in[13];
  const float* bx_b1 = (const float*)d_in[14];
  const float* wh_b1 = (const float*)d_in[15];
  const float* bh_b1 = (const float*)d_in[16];

  const size_t GXN = (size_t)SEQ * BATCH * G3H;      // 25,165,824 floats
  float* gxF = (float*)d_ws;
  float* gxB = gxF + GXN;
  char* p = (char*)(gxB + GXN);
  __bf16* xbf   = (__bf16*)p; p += (size_t)BATCH * SEQ * 512 * 2;    // x in bf16
  __bf16* y0bf  = (__bf16*)p; p += (size_t)BATCH * SEQ * 1024 * 2;   // layer0 out
  __bf16* wxf0b = (__bf16*)p; p += (size_t)G3H * 512 * 2;
  __bf16* wxb0b = (__bf16*)p; p += (size_t)G3H * 512 * 2;
  __bf16* whf0b = (__bf16*)p; p += (size_t)G3H * 512 * 2;
  __bf16* whb0b = (__bf16*)p; p += (size_t)G3H * 512 * 2;
  __bf16* wxf1b = (__bf16*)p; p += (size_t)G3H * 1024 * 2;
  __bf16* wxb1b = (__bf16*)p; p += (size_t)G3H * 1024 * 2;
  __bf16* whf1b = (__bf16*)p; p += (size_t)G3H * 512 * 2;
  __bf16* whb1b = (__bf16*)p; p += (size_t)G3H * 512 * 2;

  float* out = (float*)d_out;
  float* y1  = out;                                  // [B,S,2H]
  float* hfw = out + (size_t)BATCH * SEQ * 2 * HID;  // [2,B,H]
  float* hbw = hfw + 2 * BATCH * HID;                // [2,B,H]

  (void)hipFuncSetAttribute((const void*)gru_scan,
                            hipFuncAttributeMaxDynamicSharedMemorySize,
                            SMEM_SCAN);

  // --- precision staging: fp32 -> bf16 for x and all weight matrices ---
  const int NW  = G3H * 512;        // 786,432
  const int NW2 = G3H * 1024;       // 1,572,864
  const int NX  = BATCH * SEQ * 512;
  cvt_bf16x4<<<(NX / 4 + 255) / 256, 256, 0, stream>>>(x, xbf, NX / 4);
  cvt_bf16x4<<<(NW / 4 + 255) / 256, 256, 0, stream>>>(wx_f0, wxf0b, NW / 4);
  cvt_bf16x4<<<(NW / 4 + 255) / 256, 256, 0, stream>>>(wx_b0, wxb0b, NW / 4);
  cvt_bf16x4<<<(NW / 4 + 255) / 256, 256, 0, stream>>>(wh_f0, whf0b, NW / 4);
  cvt_bf16x4<<<(NW / 4 + 255) / 256, 256, 0, stream>>>(wh_b0, whb0b, NW / 4);
  cvt_bf16x4<<<(NW2 / 4 + 255) / 256, 256, 0, stream>>>(wx_f1, wxf1b, NW2 / 4);
  cvt_bf16x4<<<(NW2 / 4 + 255) / 256, 256, 0, stream>>>(wx_b1, wxb1b, NW2 / 4);
  cvt_bf16x4<<<(NW / 4 + 255) / 256, 256, 0, stream>>>(wh_f1, whf1b, NW / 4);
  cvt_bf16x4<<<(NW / 4 + 255) / 256, 256, 0, stream>>>(wh_b1, whb1b, NW / 4);

  dim3 gg(G3H / 64, (SEQ * BATCH) / 128);            // 24 x 128

  // layer 0
  gates_gemm<<<gg, 256, 0, stream>>>(xbf, 512, wxf0b, bx_f0, gxF);
  gates_gemm<<<gg, 256, 0, stream>>>(xbf, 512, wxb0b, bx_b0, gxB);
  gru_scan<<<2, 1024, SMEM_SCAN, stream>>>(gxF, gxB, whf0b, bh_f0, whb0b, bh_b0,
                                           nullptr, y0bf, hfw + 0, hbw + 0);
  // layer 1 (input dim 2H = 1024)
  gates_gemm<<<gg, 256, 0, stream>>>(y0bf, 1024, wxf1b, bx_f1, gxF);
  gates_gemm<<<gg, 256, 0, stream>>>(y0bf, 1024, wxb1b, bx_b1, gxB);
  gru_scan<<<2, 1024, SMEM_SCAN, stream>>>(gxF, gxB, whf1b, bh_f1, whb1b, bh_b1,
                                           y1, nullptr, hfw + BATCH * HID, hbw + BATCH * HID);
}
